// MyMultiHeadAttention_52785148068567
// MI455X (gfx1250) — matrix operand order repaired and checked by reading, not compiled, not run on hardware
//
#include <hip/hip_runtime.h>
#include <hip/hip_bf16.h>
#include <math.h>

typedef __bf16 bf16;
typedef __attribute__((ext_vector_type(8)))  bf16  bf16x8;
typedef __attribute__((ext_vector_type(16))) bf16  bf16x16;
typedef __attribute__((ext_vector_type(8)))  float f32x8;
typedef __attribute__((ext_vector_type(4)))  int   i32x4;

#define B_  4
#define S_  2048
#define D_  256
#define H_  8
#define HD_ 32
#define M_  (B_ * S_)            // 8192 rows
#define NEGBIG (-1.0e30f)
__device__ __constant__ float INV_SCALE = 0.17677669529663687f; // 1/sqrt(32)

// ---------------------------------------------------------------------------
// fp32 -> bf16 convert (weights / small tensors)
// ---------------------------------------------------------------------------
__global__ void cvt_bf16_kernel(const float* __restrict__ src, bf16* __restrict__ dst, int n) {
    int i = blockIdx.x * 256 + threadIdx.x;
    if (i < n) dst[i] = (bf16)src[i];
}

// ---------------------------------------------------------------------------
// LayerNorm: one block per row of 256. Emits bf16 xn and bf16 residual copy.
// ---------------------------------------------------------------------------
__global__ void ln_kernel(const float* __restrict__ x,
                          const float* __restrict__ gamma,
                          const float* __restrict__ beta,
                          bf16* __restrict__ xn_bf,
                          bf16* __restrict__ res_bf) {
    __shared__ float sred[256];
    const int row = blockIdx.x;
    const int tid = threadIdx.x;
    const size_t idx = (size_t)row * D_ + tid;
    const float v = x[idx];

    sred[tid] = v; __syncthreads();
    for (int off = 128; off > 0; off >>= 1) {
        if (tid < off) sred[tid] += sred[tid + off];
        __syncthreads();
    }
    const float mu = sred[0] * (1.0f / D_);
    __syncthreads();
    const float dv = v - mu;
    sred[tid] = dv * dv; __syncthreads();
    for (int off = 128; off > 0; off >>= 1) {
        if (tid < off) sred[tid] += sred[tid + off];
        __syncthreads();
    }
    const float var = sred[0] * (1.0f / D_);
    const float xn = dv * rsqrtf(var + 1e-5f) * gamma[tid] + beta[tid];
    xn_bf[idx]  = (bf16)xn;
    res_bf[idx] = (bf16)v;
}

// ---------------------------------------------------------------------------
// Core: one wave computes a 16(M) x 64(N) tile, K-loop in steps of 32.
// A: row-major [M,K] bf16.  W: row-major [N,K] bf16 (i.e. computes A @ W^T).
// A-operand layout: lane holds row (lane&15); chunk  K = g*16..+15 (contiguous).
// B-operand layout: lane holds col (lane&15); chunk  K = g*16..+15 (contiguous).
// ---------------------------------------------------------------------------
__device__ __forceinline__ void gemm_tile_16x64(const bf16* __restrict__ A,
                                                const bf16* __restrict__ W,
                                                int K, int row0, int col0,
                                                f32x8 acc[4]) {
    const int lane = threadIdx.x & 31;
    const int g    = lane >> 4;
    const int li   = lane & 15;
    const bf16* arow = A + (size_t)(row0 + li) * K;
    for (int kk = 0; kk < K; kk += 32) {
        bf16x16 a = *(const bf16x16*)(arow + kk + g * 16);
#pragma unroll
        for (int t = 0; t < 4; ++t) {
            const bf16* wrow = W + (size_t)(col0 + t * 16 + li) * K;
            bf16x16 bm = *(const bf16x16*)(wrow + kk + g * 16);
            acc[t] = __builtin_amdgcn_wmma_f32_16x16x32_bf16(
                false, a, false, bm, (short)0, acc[t], false, false);
        }
    }
}
// C/D layout: element (m = g*8 + j, n = lane&15) lives in acc[t][j], col = col0+t*16+(lane&15).

// ---------------------------------------------------------------------------
// QKV GEMM: M=8192, N=768, K=256. Epilogue scatters into q/k (row-major [n,s,d])
// and vT (transposed [n,d,s]) bf16 buffers, head layout n = b*H + h,
// column c = h*96 + which*32 + d.
// ---------------------------------------------------------------------------
__global__ void gemm_qkv_kernel(const bf16* __restrict__ A, const bf16* __restrict__ W,
                                const float* __restrict__ bias,
                                bf16* __restrict__ q, bf16* __restrict__ k,
                                bf16* __restrict__ vT) {
    const int w = threadIdx.x >> 5;
    const int row0 = blockIdx.x * 32 + (w >> 2) * 16;
    const int col0 = blockIdx.y * 256 + (w & 3) * 64;
    f32x8 acc[4] = {};
    gemm_tile_16x64(A, W, D_, row0, col0, acc);
    const int lane = threadIdx.x & 31, g = lane >> 4, li = lane & 15;
#pragma unroll
    for (int t = 0; t < 4; ++t) {
        const int col = col0 + t * 16 + li;
        const int h = col / 96, r = col % 96, which = r >> 5, d = r & 31;
        const float bv = bias[col];
#pragma unroll
        for (int j = 0; j < 8; ++j) {
            const int m = row0 + g * 8 + j;
            const int b = m >> 11, s = m & (S_ - 1);
            const int nidx = b * H_ + h;
            const float v = acc[t][j] + bv;
            if (which == 0)      q[((size_t)nidx * S_ + s) * HD_ + d] = (bf16)v;
            else if (which == 1) k[((size_t)nidx * S_ + s) * HD_ + d] = (bf16)v;
            else                 vT[((size_t)nidx * HD_ + d) * S_ + s] = (bf16)v;
        }
    }
}

// ---------------------------------------------------------------------------
// Flash attention per wave: 16 q-rows, loop over 32-key blocks.
// Computes S^T = K·Q^T so score C-layout == P A-layout (no transpose needed).
// ---------------------------------------------------------------------------
__global__ void attn_kernel(const bf16* __restrict__ qm, const bf16* __restrict__ km,
                            const bf16* __restrict__ vT, const int* __restrict__ adj,
                            bf16* __restrict__ ctx) {
    const int w = threadIdx.x >> 5;
    const int lane = threadIdx.x & 31;
    const int g = lane >> 4, li = lane & 15;
    const int n = blockIdx.y;          // n = b*H + h
    const int b = n >> 3;
    const int qbase = blockIdx.x * 64 + w * 16;
    const int q = qbase + li;

    // Q as B-operand: col = q, contiguous K(=d) chunk g*16..+15
    const bf16x16 qop = *(const bf16x16*)(qm + ((size_t)n * S_ + q) * HD_ + g * 16);
    const bf16* krow  = km + (size_t)n * S_ * HD_;
    const bf16* vbase = vT + (size_t)n * HD_ * S_;
    const int*  adjrow = adj + ((size_t)b * S_ + q) * S_;

    float m_i = NEGBIG, l_i = 0.f;
    f32x8 o0 = {}, o1 = {};

    for (int kb0 = 0; kb0 < S_; kb0 += 32) {
        // K rows as A-operand (two 16-key tiles)
        bf16x16 a0 = *(const bf16x16*)(krow + (size_t)(kb0 + li) * HD_ + g * 16);
        bf16x16 a1 = *(const bf16x16*)(krow + (size_t)(kb0 + 16 + li) * HD_ + g * 16);
        f32x8 zc = {};
        f32x8 s0 = __builtin_amdgcn_wmma_f32_16x16x32_bf16(false, a0, false, qop, (short)0, zc, false, false);
        f32x8 s1 = __builtin_amdgcn_wmma_f32_16x16x32_bf16(false, a1, false, qop, (short)0, zc, false, false);

        // mask + scale: element (key = g*8+j | +16, q = lane&15); adj row contiguous in key
        i32x4 ma0 = *(const i32x4*)(adjrow + kb0 + g * 8);
        i32x4 ma1 = *(const i32x4*)(adjrow + kb0 + g * 8 + 4);
        i32x4 mb0 = *(const i32x4*)(adjrow + kb0 + 16 + g * 8);
        i32x4 mb1 = *(const i32x4*)(adjrow + kb0 + 16 + g * 8 + 4);
#pragma unroll
        for (int j = 0; j < 4; ++j) {
            s0[j]     = ma0[j] ? s0[j]     * INV_SCALE : NEGBIG;
            s0[j + 4] = ma1[j] ? s0[j + 4] * INV_SCALE : NEGBIG;
            s1[j]     = mb0[j] ? s1[j]     * INV_SCALE : NEGBIG;
            s1[j + 4] = mb1[j] ? s1[j + 4] * INV_SCALE : NEGBIG;
        }
        // online softmax: per-lane row = q = lane&15; combine halves with one xor-16
        float mx = s0[0];
#pragma unroll
        for (int j = 1; j < 8; ++j) mx = fmaxf(mx, s0[j]);
#pragma unroll
        for (int j = 0; j < 8; ++j) mx = fmaxf(mx, s1[j]);
        mx = fmaxf(mx, __shfl_xor(mx, 16));
        const float m_new = fmaxf(m_i, mx);
        const float alpha = __expf(m_i - m_new);
        float rs = 0.f;
        bf16x16 pa;   // A-layout: e[j] = key g*8+j, e[8+j] = key 16+g*8+j  (== C-layout of s0/s1)
#pragma unroll
        for (int j = 0; j < 8; ++j) {
            const float p0 = __expf(s0[j] - m_new);
            const float p1 = __expf(s1[j] - m_new);
            rs += p0 + p1;
            pa[j] = (bf16)p0; pa[j + 8] = (bf16)p1;
        }
        rs += __shfl_xor(rs, 16);
        l_i = l_i * alpha + rs;
        m_i = m_new;
        // rescale O: row m = g*8+j of the O tile; its alpha sits in lane (g*8+j)
#pragma unroll
        for (int j = 0; j < 8; ++j) {
            const float aj = __shfl(alpha, g * 8 + j);
            o0[j] *= aj; o1[j] *= aj;
        }
        // V as B-operand from transposed layout: col = d, keys g*8..+7 and 16+g*8..+7
        bf16x8 v0lo = *(const bf16x8*)(vbase + (size_t)li * S_ + kb0 + g * 8);
        bf16x8 v0hi = *(const bf16x8*)(vbase + (size_t)li * S_ + kb0 + 16 + g * 8);
        bf16x8 v1lo = *(const bf16x8*)(vbase + (size_t)(li + 16) * S_ + kb0 + g * 8);
        bf16x8 v1hi = *(const bf16x8*)(vbase + (size_t)(li + 16) * S_ + kb0 + 16 + g * 8);
        bf16x16 v0, v1;
#pragma unroll
        for (int i = 0; i < 8; ++i) {
            v0[i] = v0lo[i]; v0[i + 8] = v0hi[i];
            v1[i] = v1lo[i]; v1[i + 8] = v1hi[i];
        }
        o0 = __builtin_amdgcn_wmma_f32_16x16x32_bf16(false, pa, false, v0, (short)0, o0, false, false);
        o1 = __builtin_amdgcn_wmma_f32_16x16x32_bf16(false, pa, false, v1, (short)0, o1, false, false);
    }
    // epilogue with the reference's reshape(H,B,..) quirk: ctx[n%4, s, (n/4)*32 + d]
    const int bt = n & 3, ht = n >> 2;
#pragma unroll
    for (int j = 0; j < 8; ++j) {
        const float lj  = __shfl(l_i, g * 8 + j);
        const float inv = (lj > 0.f) ? 1.f / lj : 0.f;
        const int s = qbase + g * 8 + j;
        const size_t base = ((size_t)bt * S_ + s) * D_ + ht * HD_;
        ctx[base + li]      = (bf16)(o0[j] * inv);
        ctx[base + 16 + li] = (bf16)(o1[j] * inv);
    }
}

// ---------------------------------------------------------------------------
// FC: y = relu(ctx @ w_fc^T + b_fc)  -> bf16
// ---------------------------------------------------------------------------
__global__ void gemm_fc_kernel(const bf16* __restrict__ A, const bf16* __restrict__ W,
                               const float* __restrict__ bias, bf16* __restrict__ y) {
    const int w = threadIdx.x >> 5;
    const int row0 = blockIdx.x * 32 + (w >> 2) * 16;
    const int col0 = (w & 3) * 64;
    f32x8 acc[4] = {};
    gemm_tile_16x64(A, W, D_, row0, col0, acc);
    const int lane = threadIdx.x & 31, g = lane >> 4, li = lane & 15;
#pragma unroll
    for (int t = 0; t < 4; ++t) {
        const int col = col0 + t * 16 + li;
        const float bv = bias[col];
#pragma unroll
        for (int j = 0; j < 8; ++j) {
            const int m = row0 + g * 8 + j;
            y[(size_t)m * D_ + col] = (bf16)fmaxf(acc[t][j] + bv, 0.f);
        }
    }
}

// ---------------------------------------------------------------------------
// Generic accumulate GEMM: C = A@W^T  (accumulate=0) or C += A@W^T (=1)
// ---------------------------------------------------------------------------
__global__ void gemm_acc_kernel(const bf16* __restrict__ A, const bf16* __restrict__ W,
                                float* __restrict__ C, int accumulate) {
    const int w = threadIdx.x >> 5;
    const int row0 = blockIdx.x * 32 + (w >> 2) * 16;
    const int col0 = (w & 3) * 64;
    f32x8 acc[4] = {};
    gemm_tile_16x64(A, W, D_, row0, col0, acc);
    const int lane = threadIdx.x & 31, g = lane >> 4, li = lane & 15;
#pragma unroll
    for (int t = 0; t < 4; ++t) {
        const int col = col0 + t * 16 + li;
#pragma unroll
        for (int j = 0; j < 8; ++j) {
            const size_t idx = (size_t)(row0 + g * 8 + j) * D_ + col;
            C[idx] = accumulate ? (C[idx] + acc[t][j]) : acc[t][j];
        }
    }
}

// ---------------------------------------------------------------------------
// Gates: z = sigmoid(zacc + b_z) (written in place), rres = bf16(sigmoid(racc)*x)
// ---------------------------------------------------------------------------
__global__ void gates_kernel(float* __restrict__ zio, const float* __restrict__ racc,
                             const float* __restrict__ bz, const float* __restrict__ x,
                             bf16* __restrict__ rres) {
    const size_t idx = (size_t)blockIdx.x * 256 + threadIdx.x;
    const int col = threadIdx.x; // blockDim==D_
    const float z = 1.f / (1.f + __expf(-(zio[idx] + bz[col])));
    zio[idx] = z;
    const float r = 1.f / (1.f + __expf(-racc[idx]));
    rres[idx] = (bf16)(r * x[idx]);
}

// ---------------------------------------------------------------------------
// Final: acc = rres@u_g^T; h = tanh(gacc + acc); out = (1-z)*x + z*h
// ---------------------------------------------------------------------------
__global__ void gemm_final_kernel(const bf16* __restrict__ A, const bf16* __restrict__ W,
                                  const float* __restrict__ gacc, const float* __restrict__ zbuf,
                                  const float* __restrict__ x, float* __restrict__ out) {
    const int w = threadIdx.x >> 5;
    const int row0 = blockIdx.x * 32 + (w >> 2) * 16;
    const int col0 = (w & 3) * 64;
    f32x8 acc[4] = {};
    gemm_tile_16x64(A, W, D_, row0, col0, acc);
    const int lane = threadIdx.x & 31, g = lane >> 4, li = lane & 15;
#pragma unroll
    for (int t = 0; t < 4; ++t) {
        const int col = col0 + t * 16 + li;
#pragma unroll
        for (int j = 0; j < 8; ++j) {
            const size_t idx = (size_t)(row0 + g * 8 + j) * D_ + col;
            const float h = tanhf(gacc[idx] + acc[t][j]);
            const float z = zbuf[idx];
            out[idx] = (1.f - z) * x[idx] + z * h;
        }
    }
}

// ---------------------------------------------------------------------------
extern "C" void kernel_launch(void* const* d_in, const int* in_sizes, int n_in,
                              void* d_out, int out_size, void* d_ws, size_t ws_size,
                              hipStream_t stream) {
    const float* x    = (const float*)d_in[0];
    const int*   adj  = (const int*)d_in[1];
    const float* wqkv = (const float*)d_in[2];
    const float* bqkv = (const float*)d_in[3];
    const float* lng  = (const float*)d_in[4];
    const float* lnb  = (const float*)d_in[5];
    const float* wfc  = (const float*)d_in[6];
    const float* bfc  = (const float*)d_in[7];
    const float* wz   = (const float*)d_in[8];
    const float* bz   = (const float*)d_in[9];
    const float* uz   = (const float*)d_in[10];
    const float* wr   = (const float*)d_in[11];
    const float* ur   = (const float*)d_in[12];
    const float* wg   = (const float*)d_in[13];
    const float* ug   = (const float*)d_in[14];
    float* out = (float*)d_out;

    char* ws = (char*)d_ws;
    const size_t MB = 1024 * 1024;
    bf16*  xn_bf  = (bf16*)(ws + 0 * MB);   // 4 MB each below
    bf16*  res_bf = (bf16*)(ws + 4 * MB);
    bf16*  q_bf   = (bf16*)(ws + 8 * MB);
    bf16*  k_bf   = (bf16*)(ws + 12 * MB);
    bf16*  vT_bf  = (bf16*)(ws + 16 * MB);
    bf16*  ctx_bf = (bf16*)(ws + 20 * MB);
    bf16*  y_bf   = (bf16*)(ws + 24 * MB);
    bf16*  rr_bf  = (bf16*)(ws + 28 * MB);
    float* zacc   = (float*)(ws + 32 * MB); // 8 MB each
    float* racc   = (float*)(ws + 40 * MB);
    float* gacc   = (float*)(ws + 48 * MB);
    bf16* wqkv_bf = (bf16*)(ws + 56 * MB);
    bf16* wfc_bf  = wqkv_bf + 768 * 256;
    bf16* wz_bf   = wfc_bf + 65536;
    bf16* uz_bf   = wz_bf + 65536;
    bf16* wr_bf   = uz_bf + 65536;
    bf16* ur_bf   = wr_bf + 65536;
    bf16* wg_bf   = ur_bf + 65536;
    bf16* ug_bf   = wg_bf + 65536;

    auto cvt = [&](const float* s, bf16* dd, int cnt) {
        cvt_bf16_kernel<<<(cnt + 255) / 256, 256, 0, stream>>>(s, dd, cnt);
    };
    cvt(wqkv, wqkv_bf, 768 * 256);
    cvt(wfc, wfc_bf, 65536); cvt(wz, wz_bf, 65536); cvt(uz, uz_bf, 65536);
    cvt(wr, wr_bf, 65536);   cvt(ur, ur_bf, 65536); cvt(wg, wg_bf, 65536);
    cvt(ug, ug_bf, 65536);

    ln_kernel<<<M_, 256, 0, stream>>>(x, lng, lnb, xn_bf, res_bf);

    dim3 blk(256);
    gemm_qkv_kernel<<<dim3(M_ / 32, 3), blk, 0, stream>>>(xn_bf, wqkv_bf, bqkv, q_bf, k_bf, vT_bf);
    attn_kernel<<<dim3(S_ / 64, B_ * H_), dim3(128), 0, stream>>>(q_bf, k_bf, vT_bf, adj, ctx_bf);
    gemm_fc_kernel<<<dim3(M_ / 32, 1), blk, 0, stream>>>(ctx_bf, wfc_bf, bfc, y_bf);

    gemm_acc_kernel<<<dim3(M_ / 32, 1), blk, 0, stream>>>(y_bf,  wz_bf, zacc, 0);
    gemm_acc_kernel<<<dim3(M_ / 32, 1), blk, 0, stream>>>(res_bf, uz_bf, zacc, 1);
    gemm_acc_kernel<<<dim3(M_ / 32, 1), blk, 0, stream>>>(y_bf,  wr_bf, racc, 0);
    gemm_acc_kernel<<<dim3(M_ / 32, 1), blk, 0, stream>>>(res_bf, ur_bf, racc, 1);
    gemm_acc_kernel<<<dim3(M_ / 32, 1), blk, 0, stream>>>(y_bf,  wg_bf, gacc, 0);

    gates_kernel<<<M_, 256, 0, stream>>>(zacc, racc, bz, x, rr_bf);
    gemm_final_kernel<<<dim3(M_ / 32, 1), blk, 0, stream>>>(rr_bf, ug_bf, gacc, zacc, x, out);
}
